// SpatialGraphMambaBlock_18717467476154
// MI455X (gfx1250) — matrix operand order repaired and checked by reading, not compiled, not run on hardware
//
#include <hip/hip_runtime.h>
#include <hip/hip_bf16.h>
#include <math.h>

// ---------------- problem constants ----------------
#define BB     8
#define LL     512
#define DM     768
#define DI     1536
#define NROW   (BB * LL)        // 4096 rows for all big GEMMs
#define NSTATE 16
#define DTRANK 48
#define XDBLW  (DTRANK + 2 * NSTATE)   // 80
#define KSP    7
#define DCONV  4

typedef __attribute__((ext_vector_type(16))) __bf16 v16bf;
typedef __attribute__((ext_vector_type(8)))  float  v8f;

#define ACT_NONE     0
#define ACT_SOFTPLUS 1

// ---------------------------------------------------------------------------
// WMMA bf16 GEMM: C[M,N] (+)= act( A[M,K] * B + bias )
// One wave computes a 64x16 strip (4 M-subtiles of 16x16), reusing each
// B fragment for 4 v_wmma_f32_16x16x32_bf16 issues. All bounds handling is
// branch-free (clamped addresses + 0/1 scale) so fragment loads stay
// unconditional: A side = float4 (global_load_b128), B side = scalar strided.
//   - A fp32 row-major (lda); rows optionally shifted along L (rowShift) with
//     zero fill outside [0,L) — implements the temporal conv taps.
//   - B fp32 addressed as B[k*bStrideK + n*bStrideN] (covers (K,N) weights and
//     (out,in,k) conv-weight slices).
// ---------------------------------------------------------------------------
__global__ __launch_bounds__(32)
void gemm_wmma_bf16(const float* __restrict__ A, const float* __restrict__ Bw,
                    const float* __restrict__ bias, float* __restrict__ C,
                    int M, int N, int K, int lda, int ldc,
                    long bStrideK, long bStrideN,
                    int L, int rowShift, int accumulate, int act)
{
    const int tileM0 = blockIdx.x * 4;   // first of 4 M-subtiles
    const int tileN  = blockIdx.y;
    const int lane   = threadIdx.x & 31;
    const int half   = lane >> 4;        // 0 or 1
    const int mr     = lane & 15;        // A-row within subtile / B,C column
    const int nBase  = tileN * 16;

    // Per-subtile A row (with L-shift, clamped + scaled for zero fill)
    long  aRow[4];
    float rowScale[4];
#pragma unroll
    for (int s = 0; s < 4; ++s) {
        const int  m    = (tileM0 + s) * 16 + mr;
        const int  bIdx = m / L;
        const int  l    = m - bIdx * L;
        const int  l2   = l + rowShift;
        const bool ok   = (l2 >= 0) && (l2 < L);
        const int  l2c  = ok ? l2 : 0;
        aRow[s]     = (long)(bIdx * L + l2c) * (long)lda;
        rowScale[s] = ok ? 1.f : 0.f;
    }

    const float* Bn = Bw + (long)(nBase + mr) * bStrideN;

    v8f acc[4] = {};

    for (int k0 = 0; k0 < K; k0 += 32) {
        // ---- B fragment (shared by all 4 subtiles): k_local = half*8 + (i&7) + (i>>3)*16
        v16bf bv;
#pragma unroll
        for (int i = 0; i < 16; ++i) {
            const int   kb  = k0 + half * 8 + (i & 7) + (i >> 3) * 16;
            const int   kbc = (kb < K) ? kb : 0;
            const float sc  = (kb < K) ? 1.f : 0.f;
            bv[i] = (__bf16)(Bn[(long)kbc * bStrideK] * sc);
        }

        // ---- A runs: av[0..7] <- A[k0+half*8 ..+7], av[8..15] <- +16
        const int  ka0  = k0 + half * 8;
        const int  ka1  = ka0 + 16;
        const bool r0ok = (ka0 < K);        // runs are 8-aligned, K % 8 == 0
        const bool r1ok = (ka1 < K);
        const int  ka0c = r0ok ? ka0 : 0;
        const int  ka1c = r1ok ? ka1 : 0;

#pragma unroll
        for (int s = 0; s < 4; ++s) {
            const float sc0 = rowScale[s] * (r0ok ? 1.f : 0.f);
            const float sc1 = rowScale[s] * (r1ok ? 1.f : 0.f);
            const float4* p0 = (const float4*)(A + aRow[s] + ka0c);
            const float4* p1 = (const float4*)(A + aRow[s] + ka1c);
            const float4 a0 = p0[0];
            const float4 a1 = p0[1];
            const float4 a2 = p1[0];
            const float4 a3 = p1[1];
            v16bf av;
            av[0]  = (__bf16)(a0.x * sc0); av[1]  = (__bf16)(a0.y * sc0);
            av[2]  = (__bf16)(a0.z * sc0); av[3]  = (__bf16)(a0.w * sc0);
            av[4]  = (__bf16)(a1.x * sc0); av[5]  = (__bf16)(a1.y * sc0);
            av[6]  = (__bf16)(a1.z * sc0); av[7]  = (__bf16)(a1.w * sc0);
            av[8]  = (__bf16)(a2.x * sc1); av[9]  = (__bf16)(a2.y * sc1);
            av[10] = (__bf16)(a2.z * sc1); av[11] = (__bf16)(a2.w * sc1);
            av[12] = (__bf16)(a3.x * sc1); av[13] = (__bf16)(a3.y * sc1);
            av[14] = (__bf16)(a3.z * sc1); av[15] = (__bf16)(a3.w * sc1);
            acc[s] = __builtin_amdgcn_wmma_f32_16x16x32_bf16(
                         false, av, false, bv, (short)0, acc[s], false, false);
        }
    }

    // ---- epilogue
    const float bval = bias ? bias[nBase + mr] : 0.f;
#pragma unroll
    for (int s = 0; s < 4; ++s) {
#pragma unroll
        for (int j = 0; j < 8; ++j) {
            // C/D layout: VGPR j holds row (j + half*8), column = lane&15
            const int  mrow = (tileM0 + s) * 16 + j + half * 8;
            const long idx  = (long)mrow * (long)ldc + (nBase + mr);
            float v = acc[s][j] + bval;
            if (accumulate) v += C[idx];
            if (act == ACT_SOFTPLUS) v = (v > 20.f) ? v : log1pf(__expf(v));
            C[idx] = v;
        }
    }
}

// ---------------------------------------------------------------------------
// LayerNorm (wave per row, wave32 shuffle reduction); optional residual add.
// ---------------------------------------------------------------------------
__global__ __launch_bounds__(256)
void layernorm_kernel(const float* __restrict__ X, const float* __restrict__ R,
                      const float* __restrict__ w, const float* __restrict__ b,
                      float* __restrict__ Y, int rows, int cols)
{
    const int wave = threadIdx.x >> 5;
    const int lane = threadIdx.x & 31;
    const int row  = blockIdx.x * 8 + wave;
    if (row >= rows) return;

    const float* xr = X + (long)row * cols;
    const float* rr = R ? (R + (long)row * cols) : nullptr;

    float s = 0.f, ss = 0.f;
    for (int c = lane; c < cols; c += 32) {
        float v = xr[c] + (rr ? rr[c] : 0.f);
        s += v; ss += v * v;
    }
    for (int off = 16; off > 0; off >>= 1) {
        s  += __shfl_xor(s,  off, 32);
        ss += __shfl_xor(ss, off, 32);
    }
    const float mu  = s / cols;
    const float var = ss / cols - mu * mu;
    const float inv = rsqrtf(var + 1e-5f);
    for (int c = lane; c < cols; c += 32) {
        float v = xr[c] + (rr ? rr[c] : 0.f);
        Y[(long)row * cols + c] = (v - mu) * inv * w[c] + b[c];
    }
}

// ---------------------------------------------------------------------------
// adj = sigmoid(nodevec1 @ nodevec2) with zero diagonal (64x64, dot over 16)
// ---------------------------------------------------------------------------
__global__ void adj_kernel(const float* __restrict__ nv1,
                           const float* __restrict__ nv2,
                           float* __restrict__ adj)
{
    const int t = blockIdx.x * blockDim.x + threadIdx.x;
    if (t >= 64 * 64) return;
    const int i = t >> 6, j = t & 63;
    float s = 0.f;
#pragma unroll
    for (int k = 0; k < 16; ++k) s += nv1[i * 16 + k] * nv2[k * 64 + j];
    const float a = 1.f / (1.f + __expf(-s));
    adj[t] = (i == j) ? 0.f : a;
}

// ---------------------------------------------------------------------------
// Mamba depthwise causal conv (D_CONV=4) + bias + SiLU.
// xm lives in the first DI columns of xz (ld 2*DI).
// ---------------------------------------------------------------------------
__global__ __launch_bounds__(256)
void dwconv_silu_kernel(const float* __restrict__ xz,
                        const float* __restrict__ cw,
                        const float* __restrict__ cb,
                        float* __restrict__ out)
{
    const long t = (long)blockIdx.x * blockDim.x + threadIdx.x;
    const long total = (long)NROW * DI;
    if (t >= total) return;
    const int  d     = (int)(t % DI);
    const long bl    = t / DI;          // 0..NROW-1
    const int  l     = (int)(bl % LL);
    const long bbase = bl - l;          // row index of (b, 0)

    float acc = cb[d];
#pragma unroll
    for (int tt = 0; tt < DCONV; ++tt) {
        const int ls = l + tt - (DCONV - 1);
        // branch-free: clamp row, scale by validity
        const int   lsc = (ls >= 0) ? ls : 0;
        const float sc  = (ls >= 0) ? 1.f : 0.f;
        acc += cw[d * DCONV + tt] * xz[(bbase + lsc) * (2 * DI) + d] * sc;
    }
    out[t] = acc / (1.f + __expf(-acc));   // SiLU
}

// ---------------------------------------------------------------------------
// Selective scan: one thread per (b, d); 16 register states; fused
// y = (sum_n h*C + D*x) * silu(z) written per step.
// ---------------------------------------------------------------------------
__global__ __launch_bounds__(256)
void scan_kernel(const float* __restrict__ dt,   const float* __restrict__ xmc,
                 const float* __restrict__ xdbl, const float* __restrict__ xz,
                 const float* __restrict__ A_log, const float* __restrict__ Dp,
                 float* __restrict__ y)
{
    const int t = blockIdx.x * blockDim.x + threadIdx.x;
    if (t >= BB * DI) return;
    const int bidx = t / DI;
    const int d    = t % DI;

    float An[NSTATE], h[NSTATE];
#pragma unroll
    for (int n = 0; n < NSTATE; ++n) {
        An[n] = -__expf(A_log[d * NSTATE + n]);
        h[n]  = 0.f;
    }
    const float Dd = Dp[d];

    for (int l = 0; l < LL; ++l) {
        const long  row = (long)bidx * LL + l;
        const float dtv = dt[row * DI + d];
        const float xv  = xmc[row * DI + d];
        const float dbx = dtv * xv;
        const float* db = xdbl + row * XDBLW;
        float acc = 0.f;
#pragma unroll
        for (int n = 0; n < NSTATE; ++n) {
            const float dA = __expf(dtv * An[n]);
            h[n] = dA * h[n] + dbx * db[DTRANK + n];          // B part
            acc += h[n] * db[DTRANK + NSTATE + n];            // C part
        }
        const float zv   = xz[row * (2 * DI) + DI + d];
        const float silz = zv / (1.f + __expf(-zv));
        y[row * DI + d] = (acc + Dd * xv) * silz;
    }
}

// ---------------------------------------------------------------------------
// gate = sigmoid(gate_pre); fused = gate*x_gcn + (1-gate)*x_mamba
// ---------------------------------------------------------------------------
__global__ __launch_bounds__(256)
void gate_fuse_kernel(const float* __restrict__ gp, const float* __restrict__ xg,
                      const float* __restrict__ xm, float* __restrict__ fused,
                      long total)
{
    const long t = (long)blockIdx.x * blockDim.x + threadIdx.x;
    if (t >= total) return;
    const float g = 1.f / (1.f + __expf(-gp[t]));
    fused[t] = g * xg[t] + (1.f - g) * xm[t];
}

// ---------------------------------------------------------------------------
// Host launcher
// ---------------------------------------------------------------------------
extern "C" void kernel_launch(void* const* d_in, const int* in_sizes, int n_in,
                              void* d_out, int out_size, void* d_ws, size_t ws_size,
                              hipStream_t stream)
{
    const float* x        = (const float*)d_in[0];
    const float* nodevec1 = (const float*)d_in[1];
    const float* nodevec2 = (const float*)d_in[2];
    const float* norm1_w  = (const float*)d_in[3];
    const float* norm1_b  = (const float*)d_in[4];
    const float* norm2_w  = (const float*)d_in[5];
    const float* norm2_b  = (const float*)d_in[6];
    const float* gcn_w    = (const float*)d_in[7];
    const float* gcn_b    = (const float*)d_in[8];
    const float* conv_w   = (const float*)d_in[9];   // (out, in, 7)
    const float* conv_b   = (const float*)d_in[10];
    const float* gate_gcn_w   = (const float*)d_in[11];
    const float* gate_gcn_b   = (const float*)d_in[12];
    const float* gate_mamba_w = (const float*)d_in[13];
    const float* gate_mamba_b = (const float*)d_in[14];
    const float* out_w    = (const float*)d_in[15];
    const float* out_b    = (const float*)d_in[16];
    const float* m_in_w   = (const float*)d_in[17];  // (768, 3072)
    const float* m_conv_w = (const float*)d_in[18];  // (1536, 4)
    const float* m_conv_b = (const float*)d_in[19];
    const float* m_xproj_w= (const float*)d_in[20];  // (1536, 80)
    const float* m_dt_w   = (const float*)d_in[21];  // (48, 1536)
    const float* m_dt_b   = (const float*)d_in[22];
    const float* m_A_log  = (const float*)d_in[23];  // (1536, 16)
    const float* m_D      = (const float*)d_in[24];
    const float* m_out_w  = (const float*)d_in[25];  // (1536, 768)

    float* out = (float*)d_out;                       // (B,L,DM) then adj
    float* adj = out + (size_t)NROW * DM;

    // ---------------- workspace layout (floats) ----------------
    float* ws = (float*)d_ws;
    const size_t SZ_DM = (size_t)NROW * DM;    // 3,145,728
    const size_t SZ_DI = (size_t)NROW * DI;    // 6,291,456
    float* XN     = ws;                         // LN(x)  (reused for 'fused')
    float* XGCN   = XN     + SZ_DM;
    float* XZ     = XGCN   + SZ_DM;             // (NROW, 2*DI)
    float* XMC    = XZ     + 2 * SZ_DI;         // conv+silu'd xm
    float* XDBL   = XMC    + SZ_DI;             // (NROW, 80)
    float* DT     = XDBL   + (size_t)NROW * XDBLW;
    float* YB     = DT     + SZ_DI;             // scan output (gated)
    float* XMAMBA = YB     + SZ_DI;
    float* GATEP  = XMAMBA + SZ_DM;
    float* OUTLIN = GATEP  + SZ_DM;
    (void)ws_size; (void)n_in; (void)in_sizes; (void)out_size;

    const dim3 blk32(32);
    const int gridM = NROW / 64;                 // 64 strips of 64 rows

    // 1) adjacency output
    adj_kernel<<<(64 * 64 + 255) / 256, 256, 0, stream>>>(nodevec1, nodevec2, adj);

    // 2) xn = LayerNorm1(x)
    layernorm_kernel<<<NROW / 8, 256, 0, stream>>>(x, nullptr, norm1_w, norm1_b,
                                                   XN, NROW, DM);

    // 3) x_gcn = xn @ gcn_w + gcn_b
    gemm_wmma_bf16<<<dim3(gridM, DM / 16), blk32, 0, stream>>>(
        XN, gcn_w, gcn_b, XGCN, NROW, DM, DM, DM, DM,
        (long)DM, 1L, LL, 0, 0, ACT_NONE);

    // 4) x_gcn += sum_t shift_{t-3}(xn) @ conv_w[:,:,t]   (+conv_b once)
    for (int t = 0; t < KSP; ++t) {
        gemm_wmma_bf16<<<dim3(gridM, DM / 16), blk32, 0, stream>>>(
            XN, conv_w + t, (t == 0) ? conv_b : nullptr, XGCN,
            NROW, DM, DM, DM, DM,
            (long)KSP, (long)(DM * KSP), LL, t - (KSP - 1) / 2, 1, ACT_NONE);
    }

    // 5) xz = xn @ m_in_w   (N = 3072)
    gemm_wmma_bf16<<<dim3(gridM, (2 * DI) / 16), blk32, 0, stream>>>(
        XN, m_in_w, nullptr, XZ, NROW, 2 * DI, DM, DM, 2 * DI,
        (long)(2 * DI), 1L, LL, 0, 0, ACT_NONE);

    // 6) xm_conv = silu(depthwise_causal_conv(xm) + m_conv_b)
    {
        const long total = (long)NROW * DI;
        dwconv_silu_kernel<<<(unsigned)((total + 255) / 256), 256, 0, stream>>>(
            XZ, m_conv_w, m_conv_b, XMC);
    }

    // 7) x_dbl = xm_conv @ m_xproj_w  (N = 80)
    gemm_wmma_bf16<<<dim3(gridM, XDBLW / 16), blk32, 0, stream>>>(
        XMC, m_xproj_w, nullptr, XDBL, NROW, XDBLW, DI, DI, XDBLW,
        (long)XDBLW, 1L, LL, 0, 0, ACT_NONE);

    // 8) dt = softplus(x_dbl[:, :48] @ m_dt_w + m_dt_b)   (K = 48)
    gemm_wmma_bf16<<<dim3(gridM, DI / 16), blk32, 0, stream>>>(
        XDBL, m_dt_w, m_dt_b, DT, NROW, DI, DTRANK, XDBLW, DI,
        (long)DI, 1L, LL, 0, 0, ACT_SOFTPLUS);

    // 9) selective scan -> y (includes +D*x and *silu(z))
    scan_kernel<<<(BB * DI + 255) / 256, 256, 0, stream>>>(
        DT, XMC, XDBL, XZ, m_A_log, m_D, YB);

    // 10) x_mamba = y @ m_out_w
    gemm_wmma_bf16<<<dim3(gridM, DM / 16), blk32, 0, stream>>>(
        YB, m_out_w, nullptr, XMAMBA, NROW, DM, DI, DI, DM,
        (long)DM, 1L, LL, 0, 0, ACT_NONE);

    // 11) gate_pre = x_gcn @ gate_gcn_w + gate_gcn_b
    gemm_wmma_bf16<<<dim3(gridM, DM / 16), blk32, 0, stream>>>(
        XGCN, gate_gcn_w, gate_gcn_b, GATEP, NROW, DM, DM, DM, DM,
        (long)DM, 1L, LL, 0, 0, ACT_NONE);

    // 12) gate_pre += x_mamba @ gate_mamba_w + gate_mamba_b
    gemm_wmma_bf16<<<dim3(gridM, DM / 16), blk32, 0, stream>>>(
        XMAMBA, gate_mamba_w, gate_mamba_b, GATEP, NROW, DM, DM, DM, DM,
        (long)DM, 1L, LL, 0, 1, ACT_NONE);

    // 13) fused = sigmoid(gate_pre)*x_gcn + (1-sigmoid)*x_mamba   (reuse XN)
    gate_fuse_kernel<<<(unsigned)(((long)SZ_DM + 255) / 256), 256, 0, stream>>>(
        GATEP, XGCN, XMAMBA, XN, (long)SZ_DM);

    // 14) out_lin = fused @ out_w + out_b
    gemm_wmma_bf16<<<dim3(gridM, DM / 16), blk32, 0, stream>>>(
        XN, out_w, out_b, OUTLIN, NROW, DM, DM, DM, DM,
        (long)DM, 1L, LL, 0, 0, ACT_NONE);

    // 15) out = LayerNorm2(out_lin + x)
    layernorm_kernel<<<NROW / 8, 256, 0, stream>>>(OUTLIN, x, norm2_w, norm2_b,
                                                   out, NROW, DM);
}
